// HydraAttention_1185410974398
// MI455X (gfx1250) — matrix operand order repaired and checked by reading, not compiled, not run on hardware
//
#include <hip/hip_runtime.h>

typedef __attribute__((ext_vector_type(8)))  __bf16 v8bf;
typedef __attribute__((ext_vector_type(16))) __bf16 v16bf;
typedef __attribute__((ext_vector_type(8)))  float  v8f;
typedef __attribute__((ext_vector_type(4)))  int    v4i;

#define D_DIM 1024
#define NROWS 32768      // B*S = 4*8192
#define NB    3072       // 3 * D (q,k,v concatenated on N)
#define BM    128
#define BN    128
#define BK    32
#define LDA   40         // LDS row stride in halves (32 + 8 pad, 80B, 16B aligned)

// ---------------- fp32 -> bf16 convert (RNE) ----------------
__device__ __forceinline__ unsigned int f32_to_bf16_bits(float f) {
    unsigned int u = __float_as_uint(f);
    return (u + 0x7FFFu + ((u >> 16) & 1u)) >> 16;
}

__global__ void cvt_f32_bf16(const float* __restrict__ src,
                             unsigned short* __restrict__ dst, int n) {
    int i = (blockIdx.x * blockDim.x + threadIdx.x) * 4;
    if (i + 3 < n) {
        float4 f = *(const float4*)(src + i);
        uint2 o;
        o.x = f32_to_bf16_bits(f.x) | (f32_to_bf16_bits(f.y) << 16);
        o.y = f32_to_bf16_bits(f.z) | (f32_to_bf16_bits(f.w) << 16);
        *(uint2*)(dst + i) = o;
    }
}

__global__ void zero_f32(float* __restrict__ p, int n) {
    int i = blockIdx.x * blockDim.x + threadIdx.x;
    if (i < n) p[i] = 0.0f;
}

// ---------------- async global -> LDS 16B copy (CDNA5, ASYNCcnt) ----------------
__device__ __forceinline__ void async_cp16(unsigned int ldsByteOff, const void* g) {
#if __has_builtin(__builtin_amdgcn_global_load_async_to_lds_b128)
    typedef __attribute__((address_space(1))) v4i* gp_t;   // global
    typedef __attribute__((address_space(3))) v4i* lp_t;   // LDS
    __builtin_amdgcn_global_load_async_to_lds_b128(
        (gp_t)(unsigned long long)g,
        (lp_t)(unsigned long)ldsByteOff,
        0, 0);
#else
    asm volatile("global_load_async_to_lds_b128 %0, %1, off"
                 :: "v"(ldsByteOff), "v"((unsigned long long)g)
                 : "memory");
#endif
}

__device__ __forceinline__ void wait_async0() {
#if __has_builtin(__builtin_amdgcn_s_wait_asynccnt)
    __builtin_amdgcn_s_wait_asynccnt(0);
#else
    asm volatile("s_wait_asynccnt 0x0" ::: "memory");
#endif
}

// ---------------- fused QKV GEMM: [32768,1024] x [1024,3072] ----------------
__global__ __launch_bounds__(256) void qkv_gemm(
    const unsigned short* __restrict__ xb,   // bf16 [NROWS][D_DIM]
    const unsigned short* __restrict__ wb,   // bf16 [NB][D_DIM]  (rows of W == cols of W^T)
    float* __restrict__ qO, float* __restrict__ kO, float* __restrict__ vO,
    const float* __restrict__ bq, const float* __restrict__ bk, const float* __restrict__ bv)
{
    __shared__ __align__(16) unsigned short smA[2][BM * LDA];
    __shared__ __align__(16) unsigned short smB[2][BN * LDA];

    const int tid   = threadIdx.x;
    const int lane  = tid & 31;
    const int wave  = tid >> 5;
    const int waveM = wave & 3;        // 4 waves along M: 32 rows each
    const int waveN = wave >> 2;       // 2 waves along N: 64 cols each
    const int mBase = blockIdx.y * BM;
    const int nBase = blockIdx.x * BN;

    // copy assignment: per stage A and B are each 128 rows x 64B = 512 x 16B chunks.
    // thread t copies A rows (t>>2, t>>2+64) seg t&3, and the same for B.
    const int crow = tid >> 2;         // 0..63
    const int cseg = tid & 3;          // 16B segment within a 64B row

    const unsigned short* gA = xb + (size_t)(mBase + crow) * D_DIM + cseg * 8;
    const unsigned short* gB = wb + (size_t)(nBase + crow) * D_DIM + cseg * 8;
    const unsigned int ldsA0 = (unsigned int)(uintptr_t)&smA[0][crow * LDA + cseg * 8];
    const unsigned int ldsA1 = (unsigned int)(uintptr_t)&smA[0][(crow + 64) * LDA + cseg * 8];
    const unsigned int ldsB0 = (unsigned int)(uintptr_t)&smB[0][crow * LDA + cseg * 8];
    const unsigned int ldsB1 = (unsigned int)(uintptr_t)&smB[0][(crow + 64) * LDA + cseg * 8];
    const unsigned int stageBytes = (unsigned int)(BM * LDA * sizeof(unsigned short));

    auto issue_stage = [&](int buf, int k0) {
        const unsigned int so = buf ? stageBytes : 0u;
        async_cp16(ldsA0 + so, gA + k0);
        async_cp16(ldsA1 + so, gA + (size_t)64 * D_DIM + k0);
        async_cp16(ldsB0 + so, gB + k0);
        async_cp16(ldsB1 + so, gB + (size_t)64 * D_DIM + k0);
    };

    v8f acc[2][4];
#pragma unroll
    for (int i = 0; i < 2; ++i)
#pragma unroll
        for (int j = 0; j < 4; ++j)
#pragma unroll
            for (int e = 0; e < 8; ++e) acc[i][j][e] = 0.0f;

    const int laneM  = lane & 15;
    const int hiHalf = lane >> 4;      // lane group 0 or 1
    const int aK = hiHalf * 8;         // A frag: K halves {aK..aK+7, aK+16..aK+23}
    const int bK = hiHalf * 8;         // B frag: K halves {bK..bK+7, bK+16..bK+23}

    issue_stage(0, 0);
    wait_async0();
    __syncthreads();

    int buf = 0;
#pragma unroll 1
    for (int kt = 0; kt < D_DIM / BK; ++kt) {
        if (kt + 1 < D_DIM / BK) issue_stage(buf ^ 1, (kt + 1) * BK);

        v16bf a[2], b[4];
#pragma unroll
        for (int i = 0; i < 2; ++i) {
            const unsigned short* p = &smA[buf][(waveM * 32 + i * 16 + laneM) * LDA + aK];
            v8bf lo = *(const v8bf*)p;
            v8bf hi = *(const v8bf*)(p + 16);
            a[i] = __builtin_shufflevector(lo, hi, 0,1,2,3,4,5,6,7,8,9,10,11,12,13,14,15);
        }
#pragma unroll
        for (int j = 0; j < 4; ++j) {
            const unsigned short* p = &smB[buf][(waveN * 64 + j * 16 + laneM) * LDA + bK];
            v8bf lo = *(const v8bf*)p;
            v8bf hi = *(const v8bf*)(p + 16);
            b[j] = __builtin_shufflevector(lo, hi, 0,1,2,3,4,5,6,7,8,9,10,11,12,13,14,15);
        }
#pragma unroll
        for (int i = 0; i < 2; ++i)
#pragma unroll
            for (int j = 0; j < 4; ++j)
                acc[i][j] = __builtin_amdgcn_wmma_f32_16x16x32_bf16(
                    false, a[i], false, b[j], (short)0, acc[i][j], false, false);

        wait_async0();          // this wave's copies into buf^1 have landed
        __syncthreads();        // all waves' copies have landed; all reads of buf done
        buf ^= 1;
    }

    // epilogue: destination is block-uniform (BN=128 divides the 1024-col boundaries)
    const int sel = blockIdx.x >> 3;                    // 0:q 1:k 2:v
    float* dst        = (sel == 0) ? qO : (sel == 1 ? kO : vO);
    const float* bias = (sel == 0) ? bq : (sel == 1 ? bk : bv);
    const int ccBase  = (nBase & 1023) + waveN * 64;

    const int rOff = hiHalf * 8;                        // C: VGPR r -> row r + rOff, col laneM
#pragma unroll
    for (int i = 0; i < 2; ++i) {
#pragma unroll
        for (int j = 0; j < 4; ++j) {
            const int gm0 = mBase + waveM * 32 + i * 16 + rOff;
            const int cc  = ccBase + j * 16 + laneM;
            const float bv_ = bias[cc];
#pragma unroll
            for (int r = 0; r < 8; ++r)
                dst[(size_t)(gm0 + r) * D_DIM + cc] = acc[i][j][r] + bv_;
        }
    }
}

// ---------------- norms + kv reduction ----------------
__global__ __launch_bounds__(256) void reduce_kv(
    const float* __restrict__ q, const float* __restrict__ k, const float* __restrict__ v,
    float* __restrict__ qinv, float* __restrict__ kv)
{
    __shared__ float red[16];
    const int tid  = threadIdx.x;
    const int lane = tid & 31;
    const int wave = tid >> 5;
    const int rowStart = blockIdx.x * 64;    // 64 rows/block; 8192/64=128 blocks per batch
    const int batch    = rowStart >> 13;

    float acc[4] = {0.f, 0.f, 0.f, 0.f};     // thread owns cols tid, tid+256, tid+512, tid+768

    for (int r = rowStart; r < rowStart + 64; ++r) {
        const size_t base = (size_t)r * D_DIM;
        float qv[4], kk[4], vv[4];
        float sq = 0.f, sk = 0.f;
#pragma unroll
        for (int m = 0; m < 4; ++m) {
            const int c = tid + m * 256;
            qv[m] = q[base + c];
            kk[m] = k[base + c];
            vv[m] = v[base + c];
            sq += qv[m] * qv[m];
            sk += kk[m] * kk[m];
        }
#pragma unroll
        for (int off = 16; off > 0; off >>= 1) {
            sq += __shfl_xor(sq, off, 32);
            sk += __shfl_xor(sk, off, 32);
        }
        if (lane == 0) { red[wave * 2] = sq; red[wave * 2 + 1] = sk; }
        __syncthreads();
        if (tid == 0) {
            float tq = 0.f, tk = 0.f;
            for (int w = 0; w < 8; ++w) { tq += red[w * 2]; tk += red[w * 2 + 1]; }
            red[0] = tq; red[1] = tk;
            qinv[r] = rsqrtf(tq);
        }
        __syncthreads();
        const float kinv = rsqrtf(red[1]);
#pragma unroll
        for (int m = 0; m < 4; ++m) acc[m] += kk[m] * vv[m] * kinv;
        __syncthreads();
    }
#pragma unroll
    for (int m = 0; m < 4; ++m)
        atomicAdd(&kv[batch * D_DIM + tid + m * 256], acc[m]);
}

// ---------------- final scale: out = q_norm * kv ----------------
__global__ __launch_bounds__(256) void scale_out(
    float* __restrict__ out, const float* __restrict__ qinv, const float* __restrict__ kv)
{
    const int i = blockIdx.x * blockDim.x + threadIdx.x;   // float4 index
    const size_t e = (size_t)i * 4;
    const int row   = (int)(e >> 10);
    const int col   = (int)(e & 1023);
    const int batch = row >> 13;
    float4 qv  = *(float4*)(out + e);
    const float4 kvv = *(const float4*)(kv + batch * D_DIM + col);
    const float qi = qinv[row];
    qv.x *= qi * kvv.x;
    qv.y *= qi * kvv.y;
    qv.z *= qi * kvv.z;
    qv.w *= qi * kvv.w;
    *(float4*)(out + e) = qv;
}

// ---------------- launch ----------------
extern "C" void kernel_launch(void* const* d_in, const int* in_sizes, int n_in,
                              void* d_out, int out_size, void* d_ws, size_t ws_size,
                              hipStream_t stream) {
    const float* x  = (const float*)d_in[0];
    const float* Wq = (const float*)d_in[1];
    const float* bq = (const float*)d_in[2];
    const float* Wk = (const float*)d_in[3];
    const float* bk = (const float*)d_in[4];
    const float* Wv = (const float*)d_in[5];
    const float* bv = (const float*)d_in[6];

    char* ws = (char*)d_ws;
    unsigned short* xb   = (unsigned short*)ws;                        // 64 MB
    unsigned short* wb   = (unsigned short*)(ws + 67108864ull);        // 6  MB
    float*          kbuf = (float*)(ws + 73400320ull);                 // 128 MB
    float*          vbuf = (float*)(ws + 207618048ull);                // 128 MB
    float*          qinv = (float*)(ws + 341835776ull);                // 128 KB
    float*          kv   = (float*)(ws + 341966848ull);                // 16 KB
    float*          qbuf = (float*)d_out;                              // q lives in d_out

    const int NX = NROWS * D_DIM;       // 33554432
    const int NW = D_DIM * D_DIM;       // 1048576

    cvt_f32_bf16<<<NX / 1024, 256, 0, stream>>>(x,  xb, NX);
    cvt_f32_bf16<<<NW / 1024, 256, 0, stream>>>(Wq, wb,           NW);
    cvt_f32_bf16<<<NW / 1024, 256, 0, stream>>>(Wk, wb + 1048576, NW);
    cvt_f32_bf16<<<NW / 1024, 256, 0, stream>>>(Wv, wb + 2097152, NW);
    zero_f32<<<16, 256, 0, stream>>>(kv, 4 * D_DIM);

    dim3 grid(NB / BN, NROWS / BM);     // 24 x 256
    qkv_gemm<<<grid, 256, 0, stream>>>(xb, wb, qbuf, kbuf, vbuf, bq, bk, bv);

    reduce_kv<<<NROWS / 64, 256, 0, stream>>>(qbuf, kbuf, vbuf, qinv, kv);
    scale_out<<<NX / 1024, 256, 0, stream>>>(qbuf, qinv, kv);
}
